// GATNetHeadsChanged4LayersReLU_LayerNorm_31628139168043
// MI455X (gfx1250) — matrix operand order repaired and checked by reading, not compiled, not run on hardware
//
#include <hip/hip_runtime.h>
#include <hip/hip_bf16.h>

typedef __attribute__((ext_vector_type(16))) _Float16 v16h;
typedef __attribute__((ext_vector_type(8)))  float    v8f;

#define NN      8192
#define EE      524288
#define ET      (EE + NN)     // edges + self loops
#define FIN     256
#define HH      2
#define FOUT    128
#define HOUT    (HH * FOUT)   // 256
#define NEG_SLOPE 0.2f

// ---------------------------------------------------------------- utilities

__device__ __forceinline__ void edge_nodes(const int* ei, int e, int& s, int& d) {
    if (e < EE) { s = ei[e]; d = ei[EE + e]; }
    else        { s = e - EE; d = e - EE; }
}

__device__ __forceinline__ void atomicMaxF(float* addr, float v) {
    unsigned int* ua = (unsigned int*)addr;
    unsigned int old = *ua;
    while (true) {
        float f = __uint_as_float(old);
        if (f >= v) break;
        unsigned int prev = atomicCAS(ua, old, __float_as_uint(v));
        if (prev == old) break;
        old = prev;
    }
}

// ---------------------------------------------------------------- init / convert

__global__ void init_kernel(float* acc, float* emax, float* denom, int nAcc, int nNH) {
    int i = blockIdx.x * blockDim.x + threadIdx.x;
    if (i < nAcc) acc[i] = 0.0f;
    if (i < nNH) { emax[i] = -3.4e38f; denom[i] = 0.0f; }
}

__global__ void cvt_f16_kernel(const float* in, _Float16* out, int n) {
    int i = blockIdx.x * blockDim.x + threadIdx.x;
    if (i < n) out[i] = (_Float16)in[i];
}

// in: [K,N] f32 row-major  ->  out: [N,K] f16 (transposed, write-coalesced)
__global__ void cvt_f16_transpose_kernel(const float* in, _Float16* out, int K, int N) {
    int i = blockIdx.x * blockDim.x + threadIdx.x;
    if (i >= K * N) return;
    int n = i / K, k = i - n * K;
    out[i] = (_Float16)in[(size_t)k * N + n];
}

// ---------------------------------------------------------------- WMMA GEMM
// C[M,N] = A[M,K] * B[K,N] with B supplied TRANSPOSED (BT[N,K], f16).
// Each wave computes a 16x64 strip: one A fragment feeds 4 WMMAs per K-step.
// All fragment loads are contiguous 32B per lane -> global_load_b128 pairs.
__global__ void gemm_f16_wmma(const _Float16* __restrict__ A,
                              const _Float16* __restrict__ BT,
                              float* __restrict__ C, int M, int N, int K) {
    const int wave = threadIdx.x >> 5;
    const int lane = threadIdx.x & 31;
    const int tile = blockIdx.x * (blockDim.x >> 5) + wave;
    const int tilesN = N >> 6;                 // 64-wide column tiles
    const int tm = (tile / tilesN) << 4;
    const int tn = (tile % tilesN) << 6;
    if (tm >= M) return;
    const int hi  = lane >> 4;                 // 0 or 1
    const int l16 = lane & 15;

    v8f acc0 = {}, acc1 = {}, acc2 = {}, acc3 = {};
    for (int k0 = 0; k0 < K; k0 += 32) {
        // A fragment: lane = row l16, contiguous K seg hi*16..hi*16+15
        v16h a;
        const _Float16* Arow = A + (size_t)(tm + l16) * K + k0 + hi * 16;
        #pragma unroll
        for (int i = 0; i < 8; ++i) { a[i] = Arow[i]; a[8 + i] = Arow[8 + i]; }

        // B fragments from BT: lane = column (tn + t*16 + l16), contiguous K seg
        #pragma unroll
        for (int t = 0; t < 4; ++t) {
            v16h b;
            const _Float16* Bp = BT + (size_t)(tn + t * 16 + l16) * K + k0 + hi * 16;
            #pragma unroll
            for (int j = 0; j < 16; ++j) b[j] = Bp[j];
            v8f* accp = (t == 0) ? &acc0 : (t == 1) ? &acc1 : (t == 2) ? &acc2 : &acc3;
            *accp = __builtin_amdgcn_wmma_f32_16x16x32_f16(
                        false, a, false, b, (short)0, *accp, false, false);
        }
    }
    // C/D layout: VGPR r -> row hi*8+r, col l16 (per 16-wide subtile)
    #pragma unroll
    for (int r = 0; r < 8; ++r) {
        float* crow = C + (size_t)(tm + hi * 8 + r) * N + tn + l16;
        crow[0]  = acc0[r];
        crow[16] = acc1[r];
        crow[32] = acc2[r];
        crow[48] = acc3[r];
    }
}

// ---------------------------------------------------------------- attention dots
__global__ void att_kernel(const float* __restrict__ h,
                           const float* __restrict__ att_src,
                           const float* __restrict__ att_dst,
                           float* __restrict__ a_s, float* __restrict__ a_d) {
    int i = blockIdx.x * blockDim.x + threadIdx.x;   // [0, N*H)
    if (i >= NN * HH) return;
    int n = i >> 1, hd = i & 1;
    const float* hp = h + (size_t)n * HOUT + hd * FOUT;
    const float* as = att_src + hd * FOUT;
    const float* ad = att_dst + hd * FOUT;
    float s = 0.0f, d = 0.0f;
    #pragma unroll 8
    for (int f = 0; f < FOUT; ++f) { float v = hp[f]; s += v * as[f]; d += v * ad[f]; }
    a_s[i] = s; a_d[i] = d;
}

// ---------------------------------------------------------------- edge softmax passes

__global__ void edge_max_kernel(const int* __restrict__ ei,
                                const float* __restrict__ a_s,
                                const float* __restrict__ a_d,
                                float* __restrict__ emax) {
    int i = blockIdx.x * blockDim.x + threadIdx.x;
    if (i >= ET * HH) return;
    int e = i >> 1, hd = i & 1, s, d;
    edge_nodes(ei, e, s, d);
    float ev = a_s[s * 2 + hd] + a_d[d * 2 + hd];
    ev = ev > 0.0f ? ev : NEG_SLOPE * ev;
    atomicMaxF(&emax[d * 2 + hd], ev);
}

__global__ void edge_sum_kernel(const int* __restrict__ ei,
                                const float* __restrict__ a_s,
                                const float* __restrict__ a_d,
                                const float* __restrict__ emax,
                                float* __restrict__ denom) {
    int i = blockIdx.x * blockDim.x + threadIdx.x;
    if (i >= ET * HH) return;
    int e = i >> 1, hd = i & 1, s, d;
    edge_nodes(ei, e, s, d);
    float ev = a_s[s * 2 + hd] + a_d[d * 2 + hd];
    ev = ev > 0.0f ? ev : NEG_SLOPE * ev;
    atomicAdd(&denom[d * 2 + hd], expf(ev - emax[d * 2 + hd]));
}

// One wave per edge: 8 features per lane; lanes 0-15 -> head 0, 16-31 -> head 1.
__global__ void edge_aggr_kernel(const int* __restrict__ ei,
                                 const float* __restrict__ a_s,
                                 const float* __restrict__ a_d,
                                 const float* __restrict__ emax,
                                 const float* __restrict__ denom,
                                 const float* __restrict__ h,
                                 float* __restrict__ acc) {
    int wv = (blockIdx.x * blockDim.x + threadIdx.x) >> 5;
    int lane = threadIdx.x & 31;
    if (wv >= ET) return;
    int s, d;
    edge_nodes(ei, wv, s, d);
    int hd = lane >> 4;
    float ev = a_s[s * 2 + hd] + a_d[d * 2 + hd];
    ev = ev > 0.0f ? ev : NEG_SLOPE * ev;
    float alpha = expf(ev - emax[d * 2 + hd]) / (denom[d * 2 + hd] + 1e-16f);
    const float* hs = h   + (size_t)s * HOUT + lane * 8;
    float*       ap = acc + (size_t)d * HOUT + lane * 8;
    #pragma unroll
    for (int j = 0; j < 8; ++j) atomicAdd(&ap[j], hs[j] * alpha);
}

// gat = relu(acc + bias); emit f16 for the next WMMA GEMM
__global__ void bias_relu_cvt_kernel(const float* __restrict__ acc,
                                     const float* __restrict__ bias,
                                     _Float16* __restrict__ out) {
    int i = blockIdx.x * blockDim.x + threadIdx.x;
    if (i >= NN * HOUT) return;
    float v = acc[i] + bias[i & (HOUT - 1)];
    out[i] = (_Float16)(v > 0.0f ? v : 0.0f);
}

// ---------------------------------------------------------------- fused MLP tail
__global__ void tail_kernel(const float* __restrict__ tmpA,
                            const float* __restrict__ b_a, const float* __restrict__ g_a, const float* __restrict__ be_a,
                            const float* __restrict__ w1,  const float* __restrict__ b1,
                            const float* __restrict__ g1,  const float* __restrict__ be1,
                            const float* __restrict__ w2,  const float* __restrict__ b2,
                            const float* __restrict__ g2,  const float* __restrict__ be2,
                            const float* __restrict__ w3,  const float* __restrict__ b3,
                            float4* __restrict__ z4) {
    int wv   = (blockIdx.x * blockDim.x + threadIdx.x) >> 5;
    int lane = threadIdx.x & 31;
    if (wv >= NN) return;

    // ---- layer A: LN( x + b_a ) * g + be, ReLU  (128 feats, 4/lane)
    float y[4];
    const float* row = tmpA + (size_t)wv * 128;
    float sum = 0.0f;
    #pragma unroll
    for (int i = 0; i < 4; ++i) { y[i] = row[i * 32 + lane] + b_a[i * 32 + lane]; sum += y[i]; }
    #pragma unroll
    for (int o = 16; o > 0; o >>= 1) sum += __shfl_xor(sum, o, 32);
    float mu = sum * (1.0f / 128.0f);
    float var = 0.0f;
    #pragma unroll
    for (int i = 0; i < 4; ++i) { float t = y[i] - mu; var += t * t; }
    #pragma unroll
    for (int o = 16; o > 0; o >>= 1) var += __shfl_xor(var, o, 32);
    float rstd = rsqrtf(var * (1.0f / 128.0f) + 1e-5f);
    #pragma unroll
    for (int i = 0; i < 4; ++i) {
        float t = (y[i] - mu) * rstd * g_a[i * 32 + lane] + be_a[i * 32 + lane];
        y[i] = t > 0.0f ? t : 0.0f;
    }

    // ---- layer 1: 128 -> 64 via shuffle-broadcast GEMV
    float c0 = 0.0f, c1 = 0.0f;
    for (int k = 0; k < 128; ++k) {
        float yk = __shfl(y[k >> 5], k & 31, 32);
        c0 += yk * w1[k * 64 + lane];
        c1 += yk * w1[k * 64 + 32 + lane];
    }
    c0 += b1[lane]; c1 += b1[32 + lane];
    float s2 = c0 + c1;
    #pragma unroll
    for (int o = 16; o > 0; o >>= 1) s2 += __shfl_xor(s2, o, 32);
    mu = s2 * (1.0f / 64.0f);
    float v2 = (c0 - mu) * (c0 - mu) + (c1 - mu) * (c1 - mu);
    #pragma unroll
    for (int o = 16; o > 0; o >>= 1) v2 += __shfl_xor(v2, o, 32);
    rstd = rsqrtf(v2 * (1.0f / 64.0f) + 1e-5f);
    c0 = (c0 - mu) * rstd * g1[lane]      + be1[lane];      c0 = c0 > 0.0f ? c0 : 0.0f;
    c1 = (c1 - mu) * rstd * g1[32 + lane] + be1[32 + lane]; c1 = c1 > 0.0f ? c1 : 0.0f;

    // ---- layer 2: 64 -> 32
    float c2 = 0.0f;
    for (int k = 0; k < 32; ++k) c2 += __shfl(c0, k, 32) * w2[k * 32 + lane];
    for (int k = 0; k < 32; ++k) c2 += __shfl(c1, k, 32) * w2[(32 + k) * 32 + lane];
    c2 += b2[lane];
    float s3 = c2;
    #pragma unroll
    for (int o = 16; o > 0; o >>= 1) s3 += __shfl_xor(s3, o, 32);
    mu = s3 * (1.0f / 32.0f);
    float v3 = (c2 - mu) * (c2 - mu);
    #pragma unroll
    for (int o = 16; o > 0; o >>= 1) v3 += __shfl_xor(v3, o, 32);
    rstd = rsqrtf(v3 * (1.0f / 32.0f) + 1e-5f);
    c2 = (c2 - mu) * rstd * g2[lane] + be2[lane];
    c2 = c2 > 0.0f ? c2 : 0.0f;

    // ---- layer 3: 32 -> 3, lane-reduce
    float p0 = c2 * w3[lane * 3 + 0];
    float p1 = c2 * w3[lane * 3 + 1];
    float p2 = c2 * w3[lane * 3 + 2];
    #pragma unroll
    for (int o = 16; o > 0; o >>= 1) {
        p0 += __shfl_xor(p0, o, 32);
        p1 += __shfl_xor(p1, o, 32);
        p2 += __shfl_xor(p2, o, 32);
    }
    if (lane == 0) {
        float z0 = p0 + b3[0], z1 = p1 + b3[1], z2 = p2 + b3[2];
        float4 o4; o4.x = z0; o4.y = z1; o4.z = z2;
        o4.w = z0 * z0 + z1 * z1 + z2 * z2;
        z4[wv] = o4;
    }
}

// ---------------------------------------------------------------- pairwise dist
__global__ void cdist_kernel(const float4* __restrict__ z4, float* __restrict__ out) {
    int j4 = blockIdx.x * blockDim.x + threadIdx.x;  // column group of 4
    int i  = blockIdx.y;
    if (j4 >= NN / 4) return;
    float4 zi = z4[i];
    float r[4];
    #pragma unroll
    for (int t = 0; t < 4; ++t) {
        float4 zj = z4[j4 * 4 + t];
        float d2 = zi.w + zj.w - 2.0f * (zi.x * zj.x + zi.y * zj.y + zi.z * zj.z);
        d2 = d2 > 0.0f ? d2 : 0.0f;
        r[t] = d2 > 0.0f ? sqrtf(d2) : 0.0f;
    }
    float4 o4; o4.x = r[0]; o4.y = r[1]; o4.z = r[2]; o4.w = r[3];
    ((float4*)out)[(size_t)i * (NN / 4) + j4] = o4;
}

// ---------------------------------------------------------------- launch

extern "C" void kernel_launch(void* const* d_in, const int* in_sizes, int n_in,
                              void* d_out, int out_size, void* d_ws, size_t ws_size,
                              hipStream_t stream) {
    const float* x        = (const float*)d_in[0];
    const int*   ei       = (const int*)  d_in[1];
    const float* W_gat    = (const float*)d_in[2];
    const float* att_src  = (const float*)d_in[3];
    const float* att_dst  = (const float*)d_in[4];
    const float* bias_gat = (const float*)d_in[5];
    const float* w_a  = (const float*)d_in[6];
    const float* b_a  = (const float*)d_in[7];
    const float* g_a  = (const float*)d_in[8];
    const float* be_a = (const float*)d_in[9];
    const float* w1 = (const float*)d_in[10];
    const float* b1 = (const float*)d_in[11];
    const float* g1 = (const float*)d_in[12];
    const float* be1 = (const float*)d_in[13];
    const float* w2 = (const float*)d_in[14];
    const float* b2 = (const float*)d_in[15];
    const float* g2 = (const float*)d_in[16];
    const float* be2 = (const float*)d_in[17];
    const float* w3 = (const float*)d_in[18];
    const float* b3 = (const float*)d_in[19];
    float* out = (float*)d_out;

    // workspace carve-up
    char* p = (char*)d_ws;
    float*    h_f32   = (float*)p;            p += (size_t)NN * HOUT * 4;   // 8 MB
    float*    acc     = (float*)p;            p += (size_t)NN * HOUT * 4;   // 8 MB
    _Float16* x_h     = (_Float16*)p;         p += (size_t)NN * FIN * 2;    // 4 MB
    _Float16* Wt_h    = (_Float16*)p;         p += (size_t)FIN * HOUT * 2;  // 128 KB (transposed)
    _Float16* wat_h   = (_Float16*)p;         p += (size_t)HOUT * 128 * 2;  // 64 KB (transposed)
    _Float16* hrelu_h = (_Float16*)p;         p += (size_t)NN * HOUT * 2;   // 4 MB
    float*    tmpA    = (float*)p;            p += (size_t)NN * 128 * 4;    // 4 MB
    float*    a_s     = (float*)p;            p += (size_t)NN * HH * 4;
    float*    a_d     = (float*)p;            p += (size_t)NN * HH * 4;
    float*    emax    = (float*)p;            p += (size_t)NN * HH * 4;
    float*    denom   = (float*)p;            p += (size_t)NN * HH * 4;
    float4*   z4      = (float4*)p;           p += (size_t)NN * 16;

    const int T = 256;

    // 0) init accumulators / segment buffers
    init_kernel<<<(NN * HOUT + T - 1) / T, T, 0, stream>>>(acc, emax, denom, NN * HOUT, NN * HH);

    // 1) f16 conversions: x plain; weights transposed to [N,K]
    cvt_f16_kernel<<<(NN * FIN + T - 1) / T, T, 0, stream>>>(x, x_h, NN * FIN);
    cvt_f16_transpose_kernel<<<(FIN * HOUT + T - 1) / T, T, 0, stream>>>(W_gat, Wt_h, FIN, HOUT);
    cvt_f16_transpose_kernel<<<(HOUT * 128 + T - 1) / T, T, 0, stream>>>(w_a, wat_h, HOUT, 128);

    // 2) h = x @ W_gat   (M=8192, N=256, K=256) — WMMA, 16x64 per wave
    {
        int tiles = (NN / 16) * (HOUT / 64);
        gemm_f16_wmma<<<tiles / 8, T, 0, stream>>>(x_h, Wt_h, h_f32, NN, HOUT, FIN);
    }

    // 3) attention logits per node
    att_kernel<<<(NN * HH + T - 1) / T, T, 0, stream>>>(h_f32, att_src, att_dst, a_s, a_d);

    // 4-6) segment softmax + weighted aggregation over edges
    edge_max_kernel<<<(ET * HH + T - 1) / T, T, 0, stream>>>(ei, a_s, a_d, emax);
    edge_sum_kernel<<<(ET * HH + T - 1) / T, T, 0, stream>>>(ei, a_s, a_d, emax, denom);
    edge_aggr_kernel<<<((size_t)ET * 32 + T - 1) / T, T, 0, stream>>>(ei, a_s, a_d, emax, denom, h_f32, acc);

    // 7) bias + ReLU + f16 convert
    bias_relu_cvt_kernel<<<(NN * HOUT + T - 1) / T, T, 0, stream>>>(acc, bias_gat, hrelu_h);

    // 8) tmpA = relu(gat) @ w_a   (M=8192, N=128, K=256) — WMMA, 16x64 per wave
    {
        int tiles = (NN / 16) * (128 / 64);
        gemm_f16_wmma<<<tiles / 8, T, 0, stream>>>(hrelu_h, wat_h, tmpA, NN, 128, HOUT);
    }

    // 9) fused LN/MLP tail -> z and |z|^2
    tail_kernel<<<(NN * 32) / T, T, 0, stream>>>(tmpA, b_a, g_a, be_a,
                                                 w1, b1, g1, be1,
                                                 w2, b2, g2, be2,
                                                 w3, b3, z4);

    // 10) pairwise distances -> out [N,N]
    {
        dim3 grid((NN / 4 + T - 1) / T, NN);
        cdist_kernel<<<grid, T, 0, stream>>>(z4, out);
    }
    (void)in_sizes; (void)n_in; (void)out_size; (void)ws_size;
}
